// CrossAttentionBlock_27298812133995
// MI455X (gfx1250) — matrix operand order repaired and checked by reading, not compiled, not run on hardware
//
#include <hip/hip_runtime.h>
#include <hip/hip_bf16.h>

// ---------------------------------------------------------------------------
// CrossAttentionBlock for MI455X (gfx1250, wave32, WMMA bf16->f32)
//   B=4, T=2048, D_MODEL=1024, N_HEADS=16, D_HEAD=64
// Pipeline:
//   K1: Q = dec@Wq^T+bq (row-major bf16), Kt/Vt = (enc@W^T+b) transposed per
//       head to [b,h,d,t] bf16 -- 3x (8192x1024)x(1024x1024) WMMA GEMMs,
//       software-pipelined (double-buffered LDS, one barrier per k-step)
//   K2: kvT[b,h][e][d] = scale * sum_t K[t,d] V[t,e]   (bf16, 64 small GEMMs)
//   K3: att = Q @ kv per head, + residual, LayerNorm(D=1024), f32 out
// f32->bf16 conversion: one v_perm_b32 per pair (truncation) to keep VALU off
// the critical path; the GEMMs are matrix-core bound, not VALU bound.
// ---------------------------------------------------------------------------

typedef __attribute__((ext_vector_type(16))) __bf16 v16bf;
typedef __attribute__((ext_vector_type(8)))  float  v8f;

struct __align__(16) U4 { unsigned int a, b, c, d; };
struct __align__(16) F4 { float x, y, z, w; };

union FragU { v16bf bf; U4 q[2]; };
union Acc   { v8f v; float f[8]; };

// pack {hi[31:16], lo[31:16]} in a single v_perm_b32
__device__ __forceinline__ unsigned int bf16pack(float lo, float hi) {
  return __builtin_amdgcn_perm(__builtin_bit_cast(unsigned int, hi),
                               __builtin_bit_cast(unsigned int, lo),
                               0x07060302u);
}
__device__ __forceinline__ unsigned short bf16one(float x) {
  return (unsigned short)(__builtin_bit_cast(unsigned int, x) >> 16);
}

#define WMMA_BF16(A, B, C) \
  __builtin_amdgcn_wmma_f32_16x16x32_bf16(false, (A), false, (B), (short)0, (C), false, false)

// ---------------------------------------------------------------------------
// Kernel 1 helpers: stage [128x32] X-tile + [64x32] W-tile, f32 -> bf16.
// Thread t loads A-row t (32 f32) and B-row (t>>1), half (t&1) (16 f32).
// ---------------------------------------------------------------------------
struct StageRegs { U4 a0, a1, a2, a3, b0, b1; };

__device__ __forceinline__ StageRegs stage_load(const float* __restrict__ Arow,
                                                const float* __restrict__ Brow,
                                                int k0) {
  StageRegs r;
  const F4* sa = (const F4*)(Arow + k0);
  F4 f0 = sa[0], f1 = sa[1], f2 = sa[2], f3 = sa[3];
  F4 f4 = sa[4], f5 = sa[5], f6 = sa[6], f7 = sa[7];
  const F4* sb = (const F4*)(Brow + k0);
  F4 g0 = sb[0], g1 = sb[1], g2 = sb[2], g3 = sb[3];
  r.a0.a = bf16pack(f0.x, f0.y); r.a0.b = bf16pack(f0.z, f0.w);
  r.a0.c = bf16pack(f1.x, f1.y); r.a0.d = bf16pack(f1.z, f1.w);
  r.a1.a = bf16pack(f2.x, f2.y); r.a1.b = bf16pack(f2.z, f2.w);
  r.a1.c = bf16pack(f3.x, f3.y); r.a1.d = bf16pack(f3.z, f3.w);
  r.a2.a = bf16pack(f4.x, f4.y); r.a2.b = bf16pack(f4.z, f4.w);
  r.a2.c = bf16pack(f5.x, f5.y); r.a2.d = bf16pack(f5.z, f5.w);
  r.a3.a = bf16pack(f6.x, f6.y); r.a3.b = bf16pack(f6.z, f6.w);
  r.a3.c = bf16pack(f7.x, f7.y); r.a3.d = bf16pack(f7.z, f7.w);
  r.b0.a = bf16pack(g0.x, g0.y); r.b0.b = bf16pack(g0.z, g0.w);
  r.b0.c = bf16pack(g1.x, g1.y); r.b0.d = bf16pack(g1.z, g1.w);
  r.b1.a = bf16pack(g2.x, g2.y); r.b1.b = bf16pack(g2.z, g2.w);
  r.b1.c = bf16pack(g3.x, g3.y); r.b1.d = bf16pack(g3.z, g3.w);
  return r;
}

__device__ __forceinline__ void stage_store(const StageRegs& r,
                                            unsigned short* __restrict__ Abuf,
                                            unsigned short* __restrict__ Bbuf,
                                            int tid) {
  U4* da = (U4*)&Abuf[tid * 40];               // padded rows (80B) stagger banks
  da[0] = r.a0; da[1] = r.a1; da[2] = r.a2; da[3] = r.a3;
  U4* db = (U4*)&Bbuf[(tid >> 1) * 40 + (tid & 1) * 16];
  db[0] = r.b0; db[1] = r.b1;
}

__device__ __forceinline__ void compute_step(const unsigned short* __restrict__ Abuf,
                                             const unsigned short* __restrict__ Bbuf,
                                             int w, int lm, int half, Acc acc[2][4]) {
  FragU a[2];
#pragma unroll
  for (int mi = 0; mi < 2; ++mi) {
    // A fragment: lane -> row (32w + 16mi + lm), K = half*8..+7, 16+half*8..+7
    const unsigned short* ar = &Abuf[(32 * w + 16 * mi + lm) * 40];
    a[mi].q[0] = *(const U4*)(ar + half * 8);
    a[mi].q[1] = *(const U4*)(ar + 16 + half * 8);
  }
  FragU b[4];
#pragma unroll
  for (int et = 0; et < 4; ++et) {
    // B fragment: lane -> col (et*16 + lm), K = half*8..+7, 16+half*8..+7
    const unsigned short* br = &Bbuf[(et * 16 + lm) * 40];
    b[et].q[0] = *(const U4*)(br + half * 8);
    b[et].q[1] = *(const U4*)(br + 16 + half * 8);
  }
#pragma unroll
  for (int mi = 0; mi < 2; ++mi)
#pragma unroll
    for (int et = 0; et < 4; ++et)
      acc[mi][et].v = WMMA_BF16(a[mi].bf, b[et].bf, acc[mi][et].v);
}

// ---------------------------------------------------------------------------
// Kernel 1: fused QKV projection.  grid=(M/128, N/64, 3), block=128 (4 waves).
// Wave w computes rows 32w..32w+31 x 64 cols (8 WMMAs per 32-wide k-step).
// blockIdx.z: 0 -> Q (row-major out), 1 -> K (transposed), 2 -> V (transposed)
// ---------------------------------------------------------------------------
__global__ __launch_bounds__(128) void qkv_proj_kernel(
    const float* __restrict__ enc, const float* __restrict__ dec,
    const float* __restrict__ Wq, const float* __restrict__ bq,
    const float* __restrict__ Wk, const float* __restrict__ bk,
    const float* __restrict__ Wv, const float* __restrict__ bv,
    unsigned short* __restrict__ Qb, unsigned short* __restrict__ Kt,
    unsigned short* __restrict__ Vt)
{
  const int mat = blockIdx.z;
  const float* X    = (mat == 0) ? dec : enc;
  const float* W    = (mat == 0) ? Wq : (mat == 1) ? Wk : Wv;
  const float* bias = (mat == 0) ? bq : (mat == 1) ? bk : bv;

  const int m0 = blockIdx.x * 128;
  const int n0 = blockIdx.y * 64;

  __shared__ __align__(16) unsigned short Al[2][128 * 40];  // 2 x 10 KB
  __shared__ __align__(16) unsigned short Bl[2][64 * 40];   // 2 x  5 KB

  const int tid = threadIdx.x;
  const int w = tid >> 5, lane = tid & 31, lm = lane & 15, half = lane >> 4;

  Acc acc[2][4];
#pragma unroll
  for (int mi = 0; mi < 2; ++mi)
#pragma unroll
    for (int e = 0; e < 4; ++e)
#pragma unroll
      for (int j = 0; j < 8; ++j) acc[mi][e].f[j] = 0.0f;

  const float* Arow = X + (size_t)(m0 + tid) * 1024;
  const float* Brow = W + (size_t)(n0 + (tid >> 1)) * 1024 + (tid & 1) * 16;

  // --- software pipeline: one barrier per k-step, ping-pong LDS ---
  StageRegs r = stage_load(Arow, Brow, 0);
  stage_store(r, Al[0], Bl[0], tid);
  __syncthreads();

  for (int kk = 0; kk < 32; ++kk) {
    const int cur = kk & 1;
    if (kk < 31) r = stage_load(Arow, Brow, (kk + 1) * 32);   // overlap w/ WMMA
    compute_step(Al[cur], Bl[cur], w, lm, half, acc);
    if (kk < 31) {
      stage_store(r, Al[1 - cur], Bl[1 - cur], tid);
      __syncthreads();
    }
  }

  // + bias
#pragma unroll
  for (int et = 0; et < 4; ++et) {
    float bv_ = bias[n0 + et * 16 + lm];
#pragma unroll
    for (int mi = 0; mi < 2; ++mi)
#pragma unroll
      for (int rr = 0; rr < 8; ++rr) acc[mi][et].f[rr] += bv_;
  }

  if (mat == 0) {
    // Q row-major bf16 [8192][1024]
#pragma unroll
    for (int mi = 0; mi < 2; ++mi)
#pragma unroll
      for (int et = 0; et < 4; ++et) {
        const int col = n0 + et * 16 + lm;
#pragma unroll
        for (int rr = 0; rr < 8; ++rr) {
          const int row = m0 + 32 * w + 16 * mi + rr + 8 * half;
          Qb[(size_t)row * 1024 + col] = bf16one(acc[mi][et].f[rr]);
        }
      }
  } else {
    // transposed per head: [b,h,d,t] — lane's 8 rows are 8 consecutive t
    unsigned short* outT = (mat == 1) ? Kt : Vt;
    const int bidx = m0 >> 11;                  // batch (128 | 2048)
    const int h = n0 >> 6;                      // one head per N tile
#pragma unroll
    for (int mi = 0; mi < 2; ++mi) {
      const int tbase = (m0 & 2047) + 32 * w + 16 * mi + 8 * half;
#pragma unroll
      for (int et = 0; et < 4; ++et) {
        const int dch = et * 16 + lm;
        U4 o;
        o.a = bf16pack(acc[mi][et].f[0], acc[mi][et].f[1]);
        o.b = bf16pack(acc[mi][et].f[2], acc[mi][et].f[3]);
        o.c = bf16pack(acc[mi][et].f[4], acc[mi][et].f[5]);
        o.d = bf16pack(acc[mi][et].f[6], acc[mi][et].f[7]);
        *(U4*)(outT + ((size_t)((bidx * 16 + h) * 64 + dch) * 2048 + tbase)) = o;
      }
    }
  }
}

// ---------------------------------------------------------------------------
// Kernel 2: kvT[b,h][e][d] = scale * sum_t K[t,d] V[t,e].
// grid=64 (b*h), block=128 (4 waves); wave w owns d-rows 16w..16w+15.
// ---------------------------------------------------------------------------
__global__ __launch_bounds__(128) void kv_kernel(
    const unsigned short* __restrict__ Kt, const unsigned short* __restrict__ Vt,
    unsigned short* __restrict__ kvT)
{
  const int bh = blockIdx.x;
  const int tid = threadIdx.x;
  const int w = tid >> 5, lane = tid & 31, lm = lane & 15, half = lane >> 4;

  Acc acc[4];
#pragma unroll
  for (int e = 0; e < 4; ++e)
#pragma unroll
    for (int j = 0; j < 8; ++j) acc[e].f[j] = 0.0f;

  const unsigned short* Krow = Kt + (size_t)(bh * 64 + 16 * w + lm) * 2048;
  const unsigned short* Vb   = Vt + (size_t)bh * 64 * 2048;

  for (int t0 = 0; t0 < 2048; t0 += 32) {
    FragU a;                                   // A = K^T: row d, K-dim = t
    a.q[0] = *(const U4*)(Krow + t0 + half * 8);
    a.q[1] = *(const U4*)(Krow + t0 + 16 + half * 8);
    FragU bm[4];                               // B = V: col e, K-dim = t
#pragma unroll
    for (int et = 0; et < 4; ++et) {
      const unsigned short* Vr = Vb + (size_t)(et * 16 + lm) * 2048 + t0;
      bm[et].q[0] = *(const U4*)(Vr + half * 8);
      bm[et].q[1] = *(const U4*)(Vr + 16 + half * 8);
    }
#pragma unroll
    for (int et = 0; et < 4; ++et)
      acc[et].v = WMMA_BF16(a.bf, bm[et].bf, acc[et].v);
  }

  const float SC = 0.125f;                     // 1/sqrt(D_HEAD)
#pragma unroll
  for (int et = 0; et < 4; ++et) {
    const int e = et * 16 + lm;
    const int dbase = 16 * w + 8 * half;       // lane's 8 consecutive d
    U4 o;
    o.a = bf16pack(acc[et].f[0] * SC, acc[et].f[1] * SC);
    o.b = bf16pack(acc[et].f[2] * SC, acc[et].f[3] * SC);
    o.c = bf16pack(acc[et].f[4] * SC, acc[et].f[5] * SC);
    o.d = bf16pack(acc[et].f[6] * SC, acc[et].f[7] * SC);
    *(U4*)(kvT + (size_t)(bh * 64 + e) * 64 + dbase) = o;   // kvT[e][d]
  }
}

// ---------------------------------------------------------------------------
// Kernel 3: att = Q@kv per head, + residual, LayerNorm over 1024, f32 out.
// grid=512 (16 rows each), block=512 (16 waves, one head per wave).
// ---------------------------------------------------------------------------
__global__ __launch_bounds__(512) void attn_ln_kernel(
    const unsigned short* __restrict__ Qb, const unsigned short* __restrict__ kvT,
    const float* __restrict__ dec, const float* __restrict__ gamma,
    const float* __restrict__ beta, float* __restrict__ out)
{
  __shared__ float att[16 * 1024];             // 64 KB (WGP has 320 KB)

  const int tid = threadIdx.x;
  const int h = tid >> 5, lane = tid & 31, lm = lane & 15, half = lane >> 4;
  const int m0 = blockIdx.x * 16;
  const int bidx = m0 >> 11;

  Acc acc[4];
#pragma unroll
  for (int e = 0; e < 4; ++e)
#pragma unroll
    for (int j = 0; j < 8; ++j) acc[e].f[j] = 0.0f;

  const unsigned short* Qrow = Qb + (size_t)(m0 + lm) * 1024 + h * 64;
  const unsigned short* kvh  = kvT + (size_t)(bidx * 16 + h) * 64 * 64;

#pragma unroll
  for (int dd = 0; dd < 64; dd += 32) {
    FragU a;
    a.q[0] = *(const U4*)(Qrow + dd + half * 8);
    a.q[1] = *(const U4*)(Qrow + dd + 16 + half * 8);
    FragU bm[4];
#pragma unroll
    for (int et = 0; et < 4; ++et) {
      const unsigned short* kr = kvh + (size_t)(et * 16 + lm) * 64 + dd;
      bm[et].q[0] = *(const U4*)(kr + half * 8);
      bm[et].q[1] = *(const U4*)(kr + 16 + half * 8);
    }
#pragma unroll
    for (int et = 0; et < 4; ++et)
      acc[et].v = WMMA_BF16(a.bf, bm[et].bf, acc[et].v);
  }

  // residual add, stage full 16x1024 row-block into LDS
#pragma unroll
  for (int et = 0; et < 4; ++et) {
    const int col = h * 64 + et * 16 + lm;
#pragma unroll
    for (int rr = 0; rr < 8; ++rr) {
      const int rl = rr + 8 * half;
      att[rl * 1024 + col] = acc[et].f[rr] + dec[(size_t)(m0 + rl) * 1024 + col];
    }
  }
  __syncthreads();

  // LayerNorm: wave h normalizes row h (32 lanes x 32 elements)
  const float* rowp = &att[h * 1024];
  float s = 0.0f, ss = 0.0f;
#pragma unroll
  for (int j = 0; j < 32; ++j) {
    float x = rowp[lane + 32 * j];
    s += x; ss += x * x;
  }
#pragma unroll
  for (int off = 16; off >= 1; off >>= 1) {    // wave32 butterfly reduce
    s  += __shfl_xor(s, off, 32);
    ss += __shfl_xor(ss, off, 32);
  }
  const float inv = 1.0f / 1024.0f;
  const float mu = s * inv;
  const float var = ss * inv - mu * mu;
  const float rs = rsqrtf(var + 1e-5f);

  float* orow = out + (size_t)(m0 + h) * 1024;
#pragma unroll
  for (int j = 0; j < 32; ++j) {
    const int c = lane + 32 * j;
    const float x = rowp[c];
    orow[c] = (x - mu) * rs * gamma[c] + beta[c];
  }
}

// ---------------------------------------------------------------------------
extern "C" void kernel_launch(void* const* d_in, const int* in_sizes, int n_in,
                              void* d_out, int out_size, void* d_ws, size_t ws_size,
                              hipStream_t stream) {
  (void)in_sizes; (void)n_in; (void)out_size; (void)ws_size;
  const float* enc   = (const float*)d_in[0];
  const float* dec   = (const float*)d_in[1];
  const float* Wq    = (const float*)d_in[2];
  const float* bq    = (const float*)d_in[3];
  const float* Wk    = (const float*)d_in[4];
  const float* bk    = (const float*)d_in[5];
  const float* Wv    = (const float*)d_in[6];
  const float* bv    = (const float*)d_in[7];
  const float* gamma = (const float*)d_in[8];
  const float* beta  = (const float*)d_in[9];
  float* out = (float*)d_out;

  // workspace layout (bf16): Q 16MB | Kt 16MB | Vt 16MB | kvT 512KB  (~48.5MB)
  unsigned short* Qb  = (unsigned short*)d_ws;
  unsigned short* Kt  = Qb + (size_t)8192 * 1024;
  unsigned short* Vt  = Kt + (size_t)8192 * 1024;
  unsigned short* kvT = Vt + (size_t)8192 * 1024;

  qkv_proj_kernel<<<dim3(64, 16, 3), 128, 0, stream>>>(
      enc, dec, Wq, bq, Wk, bk, Wv, bv, Qb, Kt, Vt);
  kv_kernel<<<64, 128, 0, stream>>>(Kt, Vt, kvT);
  attn_ln_kernel<<<512, 512, 0, stream>>>(Qb, kvT, dec, gamma, beta, out);
}
